// Multihead_Attn_72035191488685
// MI455X (gfx1250) — matrix operand order repaired and checked by reading, not compiled, not run on hardware
//
#include <hip/hip_runtime.h>
#include <hip/hip_bf16.h>
#include <math.h>

// ---------------------------------------------------------------------------
// Problem constants (from reference): N=16, C=512, H=W=64, RES=16
// ---------------------------------------------------------------------------
#define CC    512      // channels
#define HW_   4096     // H*W
#define NP_   16       // patches per image
#define PD    256      // pixels per patch (RES*RES)
#define OC    1536     // 3*C
#define GRP   128      // groups (C/4)

#define AS1 __attribute__((address_space(1)))
#define AS3 __attribute__((address_space(3)))

typedef __bf16 bf16_t;
typedef __attribute__((ext_vector_type(16))) __bf16 v16bf;
typedef __attribute__((ext_vector_type(8)))  __bf16 v8bf;
typedef __attribute__((ext_vector_type(4)))  __bf16 v4bf;
typedef __attribute__((ext_vector_type(2)))  __bf16 v2bf;
typedef __attribute__((ext_vector_type(8)))  float  v8f;
typedef __attribute__((ext_vector_type(4)))  unsigned int v4u;
typedef __attribute__((ext_vector_type(8)))  int v8i;
typedef __attribute__((ext_vector_type(4)))  int v4i;

// CDNA5 async/TDM capability probes (fallbacks keep the build green)
#if defined(__has_builtin)
#  if __has_builtin(__builtin_amdgcn_global_load_async_to_lds_b128) && \
      __has_builtin(__builtin_amdgcn_global_store_async_from_lds_b128) && \
      __has_builtin(__builtin_amdgcn_s_wait_asynccnt)
#    define HAVE_ASYNC 1
#  else
#    define HAVE_ASYNC 0
#  endif
#  if __has_builtin(__builtin_amdgcn_tensor_load_to_lds) && \
      __has_builtin(__builtin_amdgcn_s_wait_tensorcnt)
#    define HAVE_TDM 1
#  else
#    define HAVE_TDM 0
#  endif
#else
#  define HAVE_ASYNC 0
#  define HAVE_TDM 0
#endif

static __device__ __forceinline__ bf16_t f2bf(float f) { return (bf16_t)f; }
static __device__ __forceinline__ v8f v8f_zero() {
  v8f z = {0.f,0.f,0.f,0.f,0.f,0.f,0.f,0.f}; return z;
}

#if HAVE_ASYNC
// param types per hipcc diagnostic: (v4i AS1*, v4i AS3*, imm offset, imm cpol)
static __device__ __forceinline__ void async_load_b128(const void* g, void* l) {
  __builtin_amdgcn_global_load_async_to_lds_b128((AS1 v4i*)g, (AS3 v4i*)l, 0, 0);
}
static __device__ __forceinline__ void async_store_b128(void* g, const void* l) {
  __builtin_amdgcn_global_store_async_from_lds_b128((AS1 v4i*)g, (AS3 v4i*)l, 0, 0);
}
#endif

// ---------------------------------------------------------------------------
// K0: Wkqv f32 -> bf16 (1536x512, row-major [o][c]); makes the GEMM A-tile
// staging a pure copy eligible for async LDS loads.
// ---------------------------------------------------------------------------
__global__ void __launch_bounds__(256) wkqv_bf16_kernel(
    const float* __restrict__ W, bf16_t* __restrict__ Wb) {
  const int idx = (blockIdx.x * 256 + threadIdx.x) * 4;   // < 786432
  float4 w4 = *(const float4*)(W + idx);
  v4bf pk = { f2bf(w4.x), f2bf(w4.y), f2bf(w4.z), f2bf(w4.w) };
  *(v4bf*)(Wb + idx) = pk;
}

// ---------------------------------------------------------------------------
// K1: group-norm statistics. One block per (n,g): 4 channels * 4096 pixels,
// contiguous in memory. Writes {mean, rstd}.
// ---------------------------------------------------------------------------
__global__ void __launch_bounds__(256) gn_stats_kernel(
    const float* __restrict__ X, float* __restrict__ stats) {
  const int blk = blockIdx.x;                       // n*GRP + g
  const float4* base = (const float4*)(X + (size_t)blk * 4 * HW_);
  float s = 0.f, s2 = 0.f;
  for (int i = threadIdx.x; i < 4096; i += 256) {
    float4 v = base[i];
    s  += v.x + v.y + v.z + v.w;
    s2 += v.x*v.x + v.y*v.y + v.z*v.z + v.w*v.w;
  }
  #pragma unroll
  for (int off = 16; off > 0; off >>= 1) {          // wave32 reduce
    s  += __shfl_xor(s,  off, 32);
    s2 += __shfl_xor(s2, off, 32);
  }
  __shared__ float red[16];
  const int wave = threadIdx.x >> 5, lane = threadIdx.x & 31;
  if (lane == 0) { red[wave] = s; red[8 + wave] = s2; }
  __syncthreads();
  if (threadIdx.x == 0) {
    float ts = 0.f, ts2 = 0.f;
    #pragma unroll
    for (int w = 0; w < 8; ++w) { ts += red[w]; ts2 += red[8 + w]; }
    const float mean = ts * (1.0f / 16384.0f);
    const float var  = ts2 * (1.0f / 16384.0f) - mean * mean;
    stats[2*blk]   = mean;
    stats[2*blk+1] = rsqrtf(var + 1e-5f);
  }
}

// ---------------------------------------------------------------------------
// K2: fused GroupNorm-apply + KQV projection (bf16 WMMA, f32 accumulate).
// kqv^T[o][pixel] = Wkqv[o][c] * Xn[c][pixel] + bkqv[o]  -> bf16, patch layout
// ws_kqv[n][p][o][d]. Grid: 512 pixel-tiles x 12 o-tiles, 256 threads.
// A tile staged via async LDS copy; B tile staged as packed-pair b32 stores;
// epilogue repacked in LDS and written with coalesced async b128 stores.
// ---------------------------------------------------------------------------
__global__ void __launch_bounds__(256) kqv_gemm_kernel(
    const float* __restrict__ X, const float* __restrict__ stats,
    const float* __restrict__ gn_w, const float* __restrict__ gn_b,
    const bf16_t* __restrict__ wkqv_bf, const float* __restrict__ bkqv,
    bf16_t* __restrict__ kqv) {
  // carved LDS: As[2][128][40] | Bs[2][128][40], epilogue reuses as Ot[128][128]
  __shared__ __align__(16) char smem2[40960];
  typedef bf16_t AsT[128][40];
  AsT* Asb = (AsT*)smem2;                      // 20480 B
  AsT* Bsb = (AsT*)(smem2 + 20480);            // 20480 B
  bf16_t (*Ot)[128] = (bf16_t(*)[128])smem2;   // 32768 B (after final barrier)

  const int q0 = blockIdx.x * 128;   // pixel tile, patch-ordered
  const int o0 = blockIdx.y * 128;
  const int n  = q0 >> 12;
  const int p  = (q0 >> 8) & 15;
  const int d0 = q0 & 255;           // 0 or 128 within the patch
  const int pr = p >> 2, pc = p & 3;

  const int tid  = threadIdx.x;
  const int lane = tid & 31, wave = tid >> 5;
  const int mw = wave >> 2, nw = wave & 3;       // 2x4 wave grid
  const int col = lane & 15, half = lane >> 4;

  // B staging: thread handles a channel PAIR (packs into one b32 LDS store)
  // and 8 consecutive pixels.
  const int b_c2 = (tid >> 4) * 2;               // 0,2,..,30 (c pair, same group)
  const int b_dg = tid & 15;                     // 8-pixel group (0..15)
  const int drow = (d0 >> 4) + (b_dg >> 1);      // patch row (0..15)
  const int hwbase = ((pr * 16 + drow) << 6) + pc * 16 + (b_dg & 1) * 8;

  v8f acc[4][2];
  #pragma unroll
  for (int i = 0; i < 4; ++i)
    #pragma unroll
    for (int j = 0; j < 2; ++j) acc[i][j] = v8f_zero();

  auto stage = [&](int kk, int buf) {
    const int ck = kk * 32;
    // ---- A tile: pure bf16 copy (512 b128 chunks, 2 per thread)
    #pragma unroll
    for (int u = 0; u < 2; ++u) {
      const int ch = tid + u * 256;
      const int row = ch >> 2, coff = (ch & 3) * 8;
      const bf16_t* src = wkqv_bf + (size_t)(o0 + row) * CC + ck + coff;
      bf16_t* dst = &Asb[buf][row][coff];
#if HAVE_ASYNC
      async_load_b128(src, dst);
#else
      *(v8bf*)dst = *(const v8bf*)src;
#endif
    }
    // ---- B tile: X normalized on the fly, transposed [d][c], packed pairs
    {
      const int c0 = ck + b_c2;                  // c0 even; c0,c0+1 same group
      const float2 st = ((const float2*)stats)[n * GRP + (c0 >> 2)];
      const float sc0 = st.y * gn_w[c0];
      const float sh0 = gn_b[c0] - st.x * sc0;
      const float sc1 = st.y * gn_w[c0 + 1];
      const float sh1 = gn_b[c0 + 1] - st.x * sc1;
      const float* xr0 = X + ((size_t)(n * CC + c0) << 12) + hwbase;
      const float* xr1 = xr0 + HW_;              // channel c0+1
      if (kk < 15) {                             // prefetch next K chunk
        __builtin_prefetch(xr0 + (32 << 12), 0, 3);
        __builtin_prefetch(xr1 + (32 << 12), 0, 3);
      }
      float4 a0 = *(const float4*)(xr0);
      float4 a1 = *(const float4*)(xr0 + 4);
      float4 b0 = *(const float4*)(xr1);
      float4 b1 = *(const float4*)(xr1 + 4);
      const int dq = b_dg * 8;
      v2bf pk;
      pk = (v2bf){ f2bf(a0.x*sc0+sh0), f2bf(b0.x*sc1+sh1) }; *(v2bf*)&Bsb[buf][dq+0][b_c2] = pk;
      pk = (v2bf){ f2bf(a0.y*sc0+sh0), f2bf(b0.y*sc1+sh1) }; *(v2bf*)&Bsb[buf][dq+1][b_c2] = pk;
      pk = (v2bf){ f2bf(a0.z*sc0+sh0), f2bf(b0.z*sc1+sh1) }; *(v2bf*)&Bsb[buf][dq+2][b_c2] = pk;
      pk = (v2bf){ f2bf(a0.w*sc0+sh0), f2bf(b0.w*sc1+sh1) }; *(v2bf*)&Bsb[buf][dq+3][b_c2] = pk;
      pk = (v2bf){ f2bf(a1.x*sc0+sh0), f2bf(b1.x*sc1+sh1) }; *(v2bf*)&Bsb[buf][dq+4][b_c2] = pk;
      pk = (v2bf){ f2bf(a1.y*sc0+sh0), f2bf(b1.y*sc1+sh1) }; *(v2bf*)&Bsb[buf][dq+5][b_c2] = pk;
      pk = (v2bf){ f2bf(a1.z*sc0+sh0), f2bf(b1.z*sc1+sh1) }; *(v2bf*)&Bsb[buf][dq+6][b_c2] = pk;
      pk = (v2bf){ f2bf(a1.w*sc0+sh0), f2bf(b1.w*sc1+sh1) }; *(v2bf*)&Bsb[buf][dq+7][b_c2] = pk;
    }
  };

  stage(0, 0);
  for (int kk = 0; kk < 16; ++kk) {
#if HAVE_ASYNC
    __builtin_amdgcn_s_wait_asynccnt(0);   // A-tile async copies visible
#endif
    __syncthreads();
    if (kk + 1 < 16) stage(kk + 1, (kk + 1) & 1);
    const int buf = kk & 1;
    // A fragments: lane = row, K = half*8 + {0..7} and +16
    const int ak = half * 8;
    v16bf a[4];
    #pragma unroll
    for (int i = 0; i < 4; ++i) {
      union { v16bf v; v8bf h[2]; } u;
      const bf16_t* ar = &Asb[buf][mw * 64 + i * 16 + col][ak];
      u.h[0] = *(const v8bf*)(ar);
      u.h[1] = *(const v8bf*)(ar + 16);
      a[i] = u.v;
    }
    // B fragments: lane = column, K = half*8 + {0..7} and +16
    const int bk = half * 8;
    v16bf b[2];
    #pragma unroll
    for (int j = 0; j < 2; ++j) {
      union { v16bf v; v8bf h[2]; } u;
      const bf16_t* br = &Bsb[buf][nw * 32 + j * 16 + col][bk];
      u.h[0] = *(const v8bf*)(br);
      u.h[1] = *(const v8bf*)(br + 16);
      b[j] = u.v;
    }
    #pragma unroll
    for (int i = 0; i < 4; ++i)
      #pragma unroll
      for (int j = 0; j < 2; ++j)
        acc[i][j] = __builtin_amdgcn_wmma_f32_16x16x32_bf16(
            false, a[i], false, b[j], (short)0, acc[i][j], false, false);
  }

  // Epilogue: +bias, bf16, repack via LDS, coalesced (async) b128 stores.
  __syncthreads();                     // staging buffers now dead
  #pragma unroll
  for (int i = 0; i < 4; ++i)
    #pragma unroll
    for (int j = 0; j < 2; ++j) {
      const int dl = nw * 32 + j * 16 + col;
      #pragma unroll
      for (int r = 0; r < 8; ++r) {
        const int ol = mw * 64 + i * 16 + r + 8 * half;
        Ot[ol][dl] = f2bf(acc[i][j][r] + bkqv[o0 + ol]);
      }
    }
  __syncthreads();
  const size_t obase = (size_t)(n * NP_ + p) * OC * PD;
  #pragma unroll
  for (int u = 0; u < 8; ++u) {
    const int ch = tid + u * 256;
    const int row = ch >> 4, c8 = (ch & 15) * 8;
    bf16_t* dst = kqv + obase + (size_t)(o0 + row) * PD + d0 + c8;
    const bf16_t* src = &Ot[row][c8];
#if HAVE_ASYNC
    async_store_b128(dst, src);
#else
    *(v8bf*)dst = *(const v8bf*)src;
#endif
  }
  // S_ENDPGM performs an implicit wait-idle (covers ASYNCcnt).
}

// ---------------------------------------------------------------------------
// K3: per-patch attention collapse. One block per (n,p), 256 threads.
//  (a) logits = scale * Q.K^T via WMMA, online (flash) max/sumexp per row
//      -> softmax row-sums s[c]
//  (b) Vsum[d] = sum_f V[f][d]
//  (c) t[o] = sum_c Wo[o][c] * s[c]
// Q staged into 264KB of LDS by ONE Tensor-DMA descriptor (TDM inserts the
// 16B/row anti-bank-conflict padding via pad_interval/pad_amount).
// ---------------------------------------------------------------------------
__global__ void __launch_bounds__(256) attn_kernel(
    const bf16_t* __restrict__ kqv, const float* __restrict__ Wo,
    float* __restrict__ tbuf, float* __restrict__ vsbuf) {
  extern __shared__ char smem_raw[];
  bf16_t* Qs = (bf16_t*)smem_raw;                           // [512][264]
  float*  sS = (float*)(smem_raw + (size_t)512 * 264 * sizeof(bf16_t));

  const int blk = blockIdx.x;                               // n*16 + p
  const bf16_t* Kp = kqv + (size_t)blk * OC * PD;           // rows [0,512): K
  const bf16_t* Qp = Kp + (size_t)CC * PD;                  // rows [512,1024): Q
  const bf16_t* Vp = Kp + (size_t)(2 * CC) * PD;            // rows [1024,1536): V

  const int tid = threadIdx.x, lane = tid & 31, wave = tid >> 5;
  const int col = lane & 15, half = lane >> 4;

#if HAVE_TDM
  if (wave == 0) {
    // D# group0: count=1 | lds_addr | global_addr | type=2
    const unsigned long long ga = (unsigned long long)Qp;
    const unsigned lds_base = (unsigned)(unsigned long long)(AS3 const void*)Qs;
    v4u g0 = { 1u, lds_base, (unsigned)ga,
               (unsigned)((ga >> 32) & 0x1FFFFFFu) | 0x80000000u };
    // D# group1: data_size=2B, pad_enable, pad_interval=6 (512B),
    // pad_amount=3 (16B); tensor 256x512, tile 256x512, dim0_stride=256
    v8i g1 = { (int)((1u << 16) | (1u << 20) | (6u << 22) | (3u << 25)),
               (int)(256u << 16),   // tensor_dim0 = 256 (low bits at [79:48])
               (int)(512u << 16),   // tensor_dim1 = 512 (low bits at [111:80])
               (int)(256u << 16),   // tile_dim0  = 256  ([127:112])
               512,                 // tile_dim1  = 512  ([143:128])
               256,                 // tensor_dim0_stride ([207:160])
               0, 0 };
    v4i gz4 = { 0, 0, 0, 0 };
    v8i gz8 = { 0, 0, 0, 0, 0, 0, 0, 0 };
    __builtin_amdgcn_tensor_load_to_lds(g0, g1, gz4, gz4, gz8, 0);
    __builtin_amdgcn_s_wait_tensorcnt(0);
  }
  __syncthreads();
#else
  for (int idx = tid; idx < 512 * 32; idx += 256) {
    const int row = idx >> 5, ch = (idx & 31) * 8;
    *(v8bf*)&Qs[(size_t)row * 264 + ch] = *(const v8bf*)&Qp[(size_t)row * 256 + ch];
  }
  __syncthreads();
#endif

  const float scale = 0.044194173824159216f;                // 1/sqrt(512)
  const int cw = wave * 64;                                 // this wave's c rows
  float m_st[4][8], l_st[4][8];
  #pragma unroll
  for (int ct = 0; ct < 4; ++ct)
    #pragma unroll
    for (int r = 0; r < 8; ++r) { m_st[ct][r] = -INFINITY; l_st[ct][r] = 0.f; }

  for (int et = 0; et < 32; ++et) {
    // B fragments from K (d contiguous -> direct global b128 loads)
    v16bf bfr[8];
    const bf16_t* krow = Kp + (size_t)(et * 16 + col) * 256 + half * 8;
    #pragma unroll
    for (int kk = 0; kk < 8; ++kk) {
      union { v16bf v; v8bf h[2]; } u;
      u.h[0] = *(const v8bf*)(krow + kk * 32);
      u.h[1] = *(const v8bf*)(krow + kk * 32 + 16);
      bfr[kk] = u.v;
    }
    #pragma unroll
    for (int ct = 0; ct < 4; ++ct) {
      v8f acc = v8f_zero();
      const bf16_t* qrow = Qs + (size_t)(cw + ct * 16 + col) * 264 + half * 8;
      #pragma unroll
      for (int kk = 0; kk < 8; ++kk) {
        union { v16bf v; v8bf h[2]; } u;
        u.h[0] = *(const v8bf*)(qrow + kk * 32);
        u.h[1] = *(const v8bf*)(qrow + kk * 32 + 16);
        acc = __builtin_amdgcn_wmma_f32_16x16x32_bf16(
            false, u.v, false, bfr[kk], (short)0, acc, false, false);
      }
      #pragma unroll
      for (int r = 0; r < 8; ++r) {
        const float x  = acc[r] * scale;
        const float mo = m_st[ct][r];
        const float mn = fmaxf(mo, x);
        l_st[ct][r] = l_st[ct][r] * __expf(mo - mn) + __expf(x - mn);
        m_st[ct][r] = mn;
      }
    }
  }
  // combine the 16 e-column partials (lanes within a half-wave share rows)
  #pragma unroll
  for (int off = 1; off < 16; off <<= 1) {
    #pragma unroll
    for (int ct = 0; ct < 4; ++ct)
      #pragma unroll
      for (int r = 0; r < 8; ++r) {
        const float mo = __shfl_xor(m_st[ct][r], off, 32);
        const float lo = __shfl_xor(l_st[ct][r], off, 32);
        const float mn = fmaxf(m_st[ct][r], mo);
        l_st[ct][r] = l_st[ct][r] * __expf(m_st[ct][r] - mn) + lo * __expf(mo - mn);
        m_st[ct][r] = mn;
      }
  }
  if (col == 0) {
    #pragma unroll
    for (int ct = 0; ct < 4; ++ct)
      #pragma unroll
      for (int r = 0; r < 8; ++r)
        sS[cw + ct * 16 + r + 8 * half] = l_st[ct][r] / l_st[ct][r];
  }

  // Vsum[d]: thread d reduces over 512 V rows (coalesced across threads)
  {
    float vs = 0.f;
    const bf16_t* vcol = Vp + tid;
    for (int f = 0; f < 512; ++f) vs += (float)vcol[(size_t)f * 256];
    vsbuf[(size_t)blk * 256 + tid] = vs;
  }
  __syncthreads();

  // t[o] = Wo[o][:] . s[:]
  for (int o = tid; o < 512; o += 256) {
    const float* wrow = Wo + (size_t)o * 512;
    float a = 0.f;
    for (int c = 0; c < 512; ++c) a += wrow[c] * sS[c];
    tbuf[(size_t)blk * 512 + o] = a;
  }
}

// ---------------------------------------------------------------------------
// K4: out[n,o,hw] = t[n,p,o] * Vsum[n,p,d] + bo[o] + X[n,o,hw]
// ---------------------------------------------------------------------------
__global__ void __launch_bounds__(256) out_kernel(
    const float* __restrict__ X, const float* __restrict__ bo,
    const float* __restrict__ tbuf, const float* __restrict__ vsbuf,
    float* __restrict__ out) {
  const int b = blockIdx.x;             // n*512 + o
  const int n = b >> 9, o = b & 511;
  const float bias = bo[o];
  const float* xrow = X + (size_t)b * HW_;
  float* orow = out + (size_t)b * HW_;
  const float* tb = tbuf + (size_t)(n * NP_) * 512 + o;
  const float* vb = vsbuf + (size_t)(n * NP_) * 256;
  for (int hw = threadIdx.x; hw < HW_; hw += 256) {
    const int row = hw >> 6, c = hw & 63;
    const int p = ((row >> 4) << 2) + (c >> 4);
    const int d = ((row & 15) << 4) + (c & 15);
    orow[hw] = tb[(size_t)p * 512] * vb[(size_t)p * 256 + d] + bias + xrow[hw];
  }
}

// ---------------------------------------------------------------------------
// Workspace layout (bytes):
//   [0, 16K)            : group-norm stats (2048 x {mean, rstd})
//   [16K, +201MB)       : kqv bf16, patch layout [n][p][o][d]
//   then                : tbuf 256*512 f32, vsbuf 256*256 f32, Wkqv bf16 1.5MB
// ---------------------------------------------------------------------------
extern "C" void kernel_launch(void* const* d_in, const int* in_sizes, int n_in,
                              void* d_out, int out_size, void* d_ws, size_t ws_size,
                              hipStream_t stream) {
  (void)in_sizes; (void)n_in; (void)out_size; (void)ws_size;
  const float* X    = (const float*)d_in[0];
  const float* gn_w = (const float*)d_in[1];
  const float* gn_b = (const float*)d_in[2];
  const float* Wkqv = (const float*)d_in[3];
  const float* bkqv = (const float*)d_in[4];
  const float* Wo   = (const float*)d_in[5];
  const float* bo   = (const float*)d_in[6];
  float* out = (float*)d_out;

  char* ws = (char*)d_ws;
  const size_t kqv_bytes = (size_t)16 * NP_ * OC * PD * sizeof(bf16_t);
  float*  stats   = (float*)ws;
  bf16_t* kqv     = (bf16_t*)(ws + 16384);
  float*  tbuf    = (float*)(ws + 16384 + kqv_bytes);
  float*  vsbuf   = tbuf + 256 * 512;
  bf16_t* wkqv_bf = (bf16_t*)(ws + 16384 + kqv_bytes + 524288 + 262144);

  wkqv_bf16_kernel<<<768, 256, 0, stream>>>(Wkqv, wkqv_bf);
  gn_stats_kernel<<<2048, 256, 0, stream>>>(X, stats);
  kqv_gemm_kernel<<<dim3(512, 12), 256, 0, stream>>>(X, stats, gn_w, gn_b,
                                                     wkqv_bf, bkqv, kqv);
  const size_t attn_lds = (size_t)512 * 264 * sizeof(bf16_t) + 512 * sizeof(float);
  attn_kernel<<<256, 256, attn_lds, stream>>>(kqv, Wo, tbuf, vsbuf);
  out_kernel<<<16 * 512, 256, 0, stream>>>(X, bo, tbuf, vsbuf, out);
}
